// LDS_18829136625953
// MI455X (gfx1250) — matrix operand order repaired and checked by reading, not compiled, not run on hardware
//
#include <hip/hip_runtime.h>
#include <hip/hip_bf16.h>

typedef __attribute__((ext_vector_type(8)))  float          v8f;
typedef __attribute__((ext_vector_type(16))) __bf16         v16bf;
typedef __attribute__((ext_vector_type(8)))  unsigned short u16x8;
typedef __attribute__((ext_vector_type(16))) unsigned short u16x16;

#define BATCH 4
#define SEQ   2048
#define DIN   1024
#define STATE 1024
#define DOUT  1024
#define KX    10
#define ROWS  (BATCH * SEQ)   /* 8192 */

#define KC   32
#define LDSP 40               /* padded k-stride (ushorts); 80B rows, 16B aligned */

// ---------------- bf16 split helpers (RNE via bit trick) ----------------
__device__ __forceinline__ unsigned short bf16h(float f) {
    unsigned u = __float_as_uint(f);
    u += 0x7FFFu + ((u >> 16) & 1u);
    return (unsigned short)(u >> 16);
}
__device__ __forceinline__ float bf16f(unsigned short s) {
    return __uint_as_float(((unsigned)s) << 16);
}
__device__ __forceinline__ void split2(float f, unsigned short& hi, unsigned short& lo) {
    hi = bf16h(f);
    lo = bf16h(f - bf16f(hi));
}
__device__ __forceinline__ v16bf frag16(const unsigned short* p0, const unsigned short* p1) {
    u16x8 a = *reinterpret_cast<const u16x8*>(p0);
    u16x8 b = *reinterpret_cast<const u16x8*>(p1);
    u16x16 v = __builtin_shufflevector(a, b, 0, 1, 2, 3, 4, 5, 6, 7,
                                             8, 9, 10, 11, 12, 13, 14, 15);
    return __builtin_bit_cast(v16bf, v);
}
#define WMMA_BF16(A, B, C) \
    __builtin_amdgcn_wmma_f32_16x16x32_bf16(false, (A), false, (B), (short)0, (C), false, false)

// ===========================================================================
// PATH A: pre-converted bf16 hi/lo planes (needs ~144 MB workspace)
// ===========================================================================

// ---- conversion pre-passes -------------------------------------------------
// row-major f32 -> hi/lo bf16 planes (same layout). 8 elements / thread.
__global__ __launch_bounds__(256)
void conv_plane_kernel(const float* __restrict__ S,
                       unsigned short* __restrict__ Dhi,
                       unsigned short* __restrict__ Dlo)
{
    const size_t g = (size_t)blockIdx.x * 256 + threadIdx.x;
    const float4 f0 = reinterpret_cast<const float4*>(S)[g * 2];
    const float4 f1 = reinterpret_cast<const float4*>(S)[g * 2 + 1];
    const float f[8] = {f0.x, f0.y, f0.z, f0.w, f1.x, f1.y, f1.z, f1.w};
    u16x8 h, l;
#pragma unroll
    for (int j = 0; j < 8; ++j) {
        unsigned short hh, ll;
        split2(f[j], hh, ll);
        h[j] = hh; l[j] = ll;
    }
    reinterpret_cast<u16x8*>(Dhi)[g] = h;
    reinterpret_cast<u16x8*>(Dlo)[g] = l;
}

// 1024x1024 f32 -> transposed bf16 planes D[n][k] = S[k][n]
__global__ __launch_bounds__(256)
void conv_T_kernel(const float* __restrict__ S,
                   unsigned short* __restrict__ Dhi,
                   unsigned short* __restrict__ Dlo)
{
    const int g  = blockIdx.x * 256 + threadIdx.x;   // 0..131071
    const int n  = g >> 7;
    const int k0 = (g & 127) << 3;
    u16x8 h, l;
#pragma unroll
    for (int j = 0; j < 8; ++j) {
        unsigned short hh, ll;
        split2(S[(size_t)(k0 + j) * 1024 + n], hh, ll);
        h[j] = hh; l[j] = ll;
    }
    *reinterpret_cast<u16x8*>(Dhi + (size_t)n * 1024 + k0) = h;
    *reinterpret_cast<u16x8*>(Dlo + (size_t)n * 1024 + k0) = l;
}

// M[o][i][k] -> per-tap planes MT_k[o][i] (bf16 hi/lo); contiguous 320B reads
__global__ __launch_bounds__(256)
void conv_M_kernel(const float* __restrict__ Mg,
                   unsigned short* __restrict__ MThi,
                   unsigned short* __restrict__ MTlo)
{
    const int g  = blockIdx.x * 256 + threadIdx.x;   // 0..131071
    const int o  = g >> 7;
    const int i0 = (g & 127) << 3;
    const float4* src = reinterpret_cast<const float4*>(Mg + ((size_t)o * 1024 + i0) * KX);
    float4 buf4[20];
#pragma unroll
    for (int q = 0; q < 20; ++q) buf4[q] = src[q];
    const float* mv = reinterpret_cast<const float*>(buf4);   // mv[j*10 + k]
#pragma unroll
    for (int k = 0; k < KX; ++k) {
        u16x8 h, l;
#pragma unroll
        for (int j = 0; j < 8; ++j) {
            unsigned short hh, ll;
            split2(mv[j * KX + k], hh, ll);
            h[j] = hh; l[j] = ll;
        }
        *reinterpret_cast<u16x8*>(MThi + (size_t)k * 1024 * 1024 + (size_t)o * 1024 + i0) = h;
        *reinterpret_cast<u16x8*>(MTlo + (size_t)k * 1024 * 1024 + (size_t)o * 1024 + i0) = l;
    }
}

// ---- tile machinery: block 128x64, 8 waves, wave tile 32x32 ---------------
struct SmemPre {
    unsigned short Ahi[128][LDSP];
    unsigned short Alo[128][LDSP];
    unsigned short Bhi[64][LDSP];
    unsigned short Blo[64][LDSP];
};

__device__ __forceinline__ void stage_pre_B(SmemPre& s, int tid,
        const unsigned short* __restrict__ Ghi, const unsigned short* __restrict__ Glo,
        int col0, int kbase)
{
    const int r = tid >> 2;
    const int q = (tid & 3) << 3;
    const size_t off = (size_t)(col0 + r) * 1024 + kbase + q;
    *reinterpret_cast<u16x8*>(&s.Bhi[r][q]) = *reinterpret_cast<const u16x8*>(Ghi + off);
    *reinterpret_cast<u16x8*>(&s.Blo[r][q]) = *reinterpret_cast<const u16x8*>(Glo + off);
}

__device__ __forceinline__ void tile_mma_pre(const SmemPre& s, int wm, int wn,
                                             int half, int lm, v8f acc[2][2])
{
    v16bf ah[2], al[2], bh[2], bl[2];
#pragma unroll
    for (int mi = 0; mi < 2; ++mi) {
        const int m = wm * 32 + mi * 16 + lm;
        ah[mi] = frag16(&s.Ahi[m][8 * half], &s.Ahi[m][16 + 8 * half]);
        al[mi] = frag16(&s.Alo[m][8 * half], &s.Alo[m][16 + 8 * half]);
    }
#pragma unroll
    for (int ni = 0; ni < 2; ++ni) {
        const int n = wn * 32 + ni * 16 + lm;
        bh[ni] = frag16(&s.Bhi[n][8 * half], &s.Bhi[n][16 + 8 * half]);
        bl[ni] = frag16(&s.Blo[n][8 * half], &s.Blo[n][16 + 8 * half]);
    }
#pragma unroll
    for (int mi = 0; mi < 2; ++mi)
#pragma unroll
        for (int ni = 0; ni < 2; ++ni) {
            acc[mi][ni] = WMMA_BF16(ah[mi], bh[ni], acc[mi][ni]);
            acc[mi][ni] = WMMA_BF16(ah[mi], bl[ni], acc[mi][ni]);
            acc[mi][ni] = WMMA_BF16(al[mi], bh[ni], acc[mi][ni]);
        }
}

__device__ __forceinline__ void write_tile(float* __restrict__ Og, const v8f acc[2][2],
        int row0, int col0, int wm, int wn, int half, int lm)
{
#pragma unroll
    for (int mi = 0; mi < 2; ++mi) {
        const int orow = row0 + wm * 32 + mi * 16 + half * 8;
#pragma unroll
        for (int ni = 0; ni < 2; ++ni) {
            const int oc = col0 + wn * 32 + ni * 16 + lm;
#pragma unroll
            for (int g = 0; g < 8; ++g)
                Og[(size_t)(orow + g) * 1024 + oc] = acc[mi][ni][g];
        }
    }
}

// ---- Path-A GEMM: O[8192,1024](f32) = A @ B, operands pre-split bf16 ------
__global__ __launch_bounds__(256)
void gemm_pre(const unsigned short* __restrict__ Ahi, const unsigned short* __restrict__ Alo,
              const unsigned short* __restrict__ Bhi, const unsigned short* __restrict__ Blo,
              float* __restrict__ Og)
{
    __shared__ SmemPre s;
    const int tid  = threadIdx.x;
    const int lane = tid & 31;
    const int w    = tid >> 5;
    const int wm   = w & 3;
    const int wn   = w >> 2;
    const int half = lane >> 4;
    const int lm   = lane & 15;
    const int row0 = blockIdx.y * 128;
    const int col0 = blockIdx.x * 64;

    v8f acc[2][2] = {};

    for (int kbase = 0; kbase < 1024; kbase += KC) {
#pragma unroll
        for (int c = 0; c < 2; ++c) {     // A: 128 rows x 32 k
            const int cc = tid * 2 + c;
            const int r = cc >> 2;
            const int q = (cc & 3) << 3;
            const size_t off = (size_t)(row0 + r) * 1024 + kbase + q;
            *reinterpret_cast<u16x8*>(&s.Ahi[r][q]) = *reinterpret_cast<const u16x8*>(Ahi + off);
            *reinterpret_cast<u16x8*>(&s.Alo[r][q]) = *reinterpret_cast<const u16x8*>(Alo + off);
        }
        stage_pre_B(s, tid, Bhi, Blo, col0, kbase);
        __syncthreads();
        tile_mma_pre(s, wm, wn, half, lm, acc);
        __syncthreads();
    }
    write_tile(Og, acc, row0, col0, wm, wn, half, lm);
}

// ---- scan that emits split-bf16 H directly --------------------------------
__global__ __launch_bounds__(256)
void scan_split_kernel(const float* __restrict__ V, const float* __restrict__ Adiag,
                       const float* __restrict__ h0,
                       unsigned short* __restrict__ Hhi, unsigned short* __restrict__ Hlo)
{
    const int g = blockIdx.x * 256 + threadIdx.x;   // 0 .. 4095
    const int b = g >> 10;
    const int si = g & 1023;
    const float a = Adiag[si];
    float h = h0[si];
    const float* p = V + (size_t)b * SEQ * 1024 + si;
    unsigned short* ph = Hhi + (size_t)b * SEQ * 1024 + si;
    unsigned short* pl = Hlo + (size_t)b * SEQ * 1024 + si;

    for (int t = 0; t < SEQ; t += 8) {
        float v[8];
#pragma unroll
        for (int j = 0; j < 8; ++j)
            v[j] = p[(size_t)(t + j) * 1024];
#pragma unroll
        for (int j = 0; j < 8; ++j) {
            h = fmaf(a, h, v[j]);
            unsigned short hh, ll;
            split2(h, hh, ll);
            ph[(size_t)(t + j) * 1024] = hh;
            pl[(size_t)(t + j) * 1024] = ll;
        }
    }
}

// ---- Path-A fused epilogue: out = H@C + sum_k shift_k(X)@MT_k -------------
__global__ __launch_bounds__(256)
void fused_pre(const unsigned short* __restrict__ Hhi, const unsigned short* __restrict__ Hlo,
               const unsigned short* __restrict__ CThi, const unsigned short* __restrict__ CTlo,
               const unsigned short* __restrict__ Xhi, const unsigned short* __restrict__ Xlo,
               const unsigned short* __restrict__ MThi, const unsigned short* __restrict__ MTlo,
               float* __restrict__ Og)
{
    __shared__ SmemPre s;
    const int tid  = threadIdx.x;
    const int lane = tid & 31;
    const int w    = tid >> 5;
    const int wm   = w & 3;
    const int wn   = w >> 2;
    const int half = lane >> 4;
    const int lm   = lane & 15;
    const int row0 = blockIdx.y * 128;
    const int col0 = blockIdx.x * 64;
    const int bidx = row0 / SEQ;       // tiles never straddle batches (2048 % 128 == 0)
    const int t0   = row0 % SEQ;

    v8f acc[2][2] = {};

    for (int p = 0; p < 1 + KX; ++p) {
        const int shift = (p == 0) ? 0 : (p - 1);
        const unsigned short* Ah = (p == 0) ? Hhi : Xhi;
        const unsigned short* Al = (p == 0) ? Hlo : Xlo;
        const unsigned short* Bh = (p == 0) ? CThi : (MThi + (size_t)(p - 1) * 1024 * 1024);
        const unsigned short* Bl = (p == 0) ? CTlo : (MTlo + (size_t)(p - 1) * 1024 * 1024);

        for (int kbase = 0; kbase < 1024; kbase += KC) {
#pragma unroll
            for (int c = 0; c < 2; ++c) {     // A with causal shift
                const int cc = tid * 2 + c;
                const int r = cc >> 2;
                const int q = (cc & 3) << 3;
                const int trow = t0 + r - shift;
                u16x8 h = {}, l = {};
                if (trow >= 0) {
                    const size_t off = ((size_t)bidx * SEQ + trow) * 1024 + kbase + q;
                    h = *reinterpret_cast<const u16x8*>(Ah + off);
                    l = *reinterpret_cast<const u16x8*>(Al + off);
                }
                *reinterpret_cast<u16x8*>(&s.Ahi[r][q]) = h;
                *reinterpret_cast<u16x8*>(&s.Alo[r][q]) = l;
            }
            stage_pre_B(s, tid, Bh, Bl, col0, kbase);
            __syncthreads();
            tile_mma_pre(s, wm, wn, half, lm, acc);
            __syncthreads();
        }
    }
    write_tile(Og, acc, row0, col0, wm, wn, half, lm);
}

// ===========================================================================
// PATH B fallback: convert-in-kernel (needs only 72 MB workspace)
// ===========================================================================
#define BM 64
#define BN 64

__global__ __launch_bounds__(256)
void transpose_M_kernel(const float* __restrict__ Mg, float* __restrict__ MT)
{
    const int g = blockIdx.x * 256 + threadIdx.x;
    const int o = g & 1023;
    const int i = g >> 10;
    const float* src = Mg + ((size_t)o * 1024 + (size_t)i) * KX;
#pragma unroll
    for (int k = 0; k < KX; ++k)
        MT[(size_t)k * (1024 * 1024) + (size_t)i * 1024 + o] = src[k];
}

struct TileSmem {
    unsigned short Ahi[BM][LDSP];
    unsigned short Alo[BM][LDSP];
    unsigned short Bth[BN][LDSP];
    unsigned short Btl[BN][LDSP];
};

__device__ __forceinline__ void stage_A_row(TileSmem& s, int r, int c,
                                            const float4 f0, const float4 f1)
{
    split2(f0.x, s.Ahi[r][c + 0], s.Alo[r][c + 0]);
    split2(f0.y, s.Ahi[r][c + 1], s.Alo[r][c + 1]);
    split2(f0.z, s.Ahi[r][c + 2], s.Alo[r][c + 2]);
    split2(f0.w, s.Ahi[r][c + 3], s.Alo[r][c + 3]);
    split2(f1.x, s.Ahi[r][c + 4], s.Alo[r][c + 4]);
    split2(f1.y, s.Ahi[r][c + 5], s.Alo[r][c + 5]);
    split2(f1.z, s.Ahi[r][c + 6], s.Alo[r][c + 6]);
    split2(f1.w, s.Ahi[r][c + 7], s.Alo[r][c + 7]);
}
__device__ __forceinline__ void stage_B_row(TileSmem& s, int kk, int c,
                                            const float4 f0, const float4 f1)
{
    split2(f0.x, s.Bth[c + 0][kk], s.Btl[c + 0][kk]);
    split2(f0.y, s.Bth[c + 1][kk], s.Btl[c + 1][kk]);
    split2(f0.z, s.Bth[c + 2][kk], s.Btl[c + 2][kk]);
    split2(f0.w, s.Bth[c + 3][kk], s.Btl[c + 3][kk]);
    split2(f1.x, s.Bth[c + 4][kk], s.Btl[c + 4][kk]);
    split2(f1.y, s.Bth[c + 5][kk], s.Btl[c + 5][kk]);
    split2(f1.z, s.Bth[c + 6][kk], s.Btl[c + 6][kk]);
    split2(f1.w, s.Bth[c + 7][kk], s.Btl[c + 7][kk]);
}

__device__ __forceinline__ void tile_mma(const TileSmem& s, int wm, int wn,
                                         int half, int lm, v8f& acc0, v8f& acc1)
{
    const int m = wm * 16 + lm;
    const v16bf ah = frag16(&s.Ahi[m][8 * half], &s.Ahi[m][16 + 8 * half]);
    const v16bf al = frag16(&s.Alo[m][8 * half], &s.Alo[m][16 + 8 * half]);
    const int n0 = wn * 32 + lm;
    const int n1 = n0 + 16;
    const v16bf bh0 = frag16(&s.Bth[n0][8 * half], &s.Bth[n0][16 + 8 * half]);
    const v16bf bl0 = frag16(&s.Btl[n0][8 * half], &s.Btl[n0][16 + 8 * half]);
    const v16bf bh1 = frag16(&s.Bth[n1][8 * half], &s.Bth[n1][16 + 8 * half]);
    const v16bf bl1 = frag16(&s.Btl[n1][8 * half], &s.Btl[n1][16 + 8 * half]);
    acc0 = WMMA_BF16(ah, bh0, acc0);
    acc1 = WMMA_BF16(ah, bh1, acc1);
    acc0 = WMMA_BF16(ah, bl0, acc0);
    acc1 = WMMA_BF16(ah, bl1, acc1);
    acc0 = WMMA_BF16(al, bh0, acc0);
    acc1 = WMMA_BF16(al, bh1, acc1);
}

__global__ __launch_bounds__(256)
void gemm_bf16s_wmma(const float* __restrict__ Ag, const float* __restrict__ Bg,
                     float* __restrict__ Og)
{
    __shared__ TileSmem s;
    const int tid  = threadIdx.x;
    const int lane = tid & 31;
    const int w    = tid >> 5;
    const int wm   = w & 3;
    const int wn   = w >> 2;
    const int half = lane >> 4;
    const int lm   = lane & 15;
    const int row0 = blockIdx.y * BM;
    const int col0 = blockIdx.x * BN;

    v8f acc0 = {}, acc1 = {};

    for (int kbase = 0; kbase < 1024; kbase += KC) {
        {
            const int r = tid >> 2;
            const int c = (tid & 3) << 3;
            const float* src = Ag + (size_t)(row0 + r) * 1024 + kbase + c;
            stage_A_row(s, r, c, *reinterpret_cast<const float4*>(src),
                                  *reinterpret_cast<const float4*>(src + 4));
        }
        {
            const int kk = tid >> 3;
            const int c  = (tid & 7) << 3;
            const float* src = Bg + (size_t)(kbase + kk) * 1024 + col0 + c;
            stage_B_row(s, kk, c, *reinterpret_cast<const float4*>(src),
                                   *reinterpret_cast<const float4*>(src + 4));
        }
        __syncthreads();
        tile_mma(s, wm, wn, half, lm, acc0, acc1);
        __syncthreads();
    }

    const int orow = row0 + wm * 16 + half * 8;
    const int oc0  = col0 + wn * 32 + lm;
#pragma unroll
    for (int g = 0; g < 8; ++g) {
        Og[(size_t)(orow + g) * 1024 + oc0]      = acc0[g];
        Og[(size_t)(orow + g) * 1024 + oc0 + 16] = acc1[g];
    }
}

__global__ __launch_bounds__(256)
void scan_kernel(float* __restrict__ V, const float* __restrict__ Adiag,
                 const float* __restrict__ h0)
{
    const int g = blockIdx.x * 256 + threadIdx.x;
    const int b = g >> 10;
    const int si = g & 1023;
    const float a = Adiag[si];
    float h = h0[si];
    float* p = V + (size_t)b * SEQ * 1024 + si;

    for (int t = 0; t < SEQ; t += 8) {
        float v[8];
#pragma unroll
        for (int j = 0; j < 8; ++j)
            v[j] = p[(size_t)(t + j) * 1024];
#pragma unroll
        for (int j = 0; j < 8; ++j) {
            h = fmaf(a, h, v[j]);
            p[(size_t)(t + j) * 1024] = h;
        }
    }
}

__global__ __launch_bounds__(256)
void fused_out_wmma(const float* __restrict__ H, const float* __restrict__ Cg,
                    const float* __restrict__ X, const float* __restrict__ MT,
                    float* __restrict__ Og)
{
    __shared__ TileSmem s;
    const int tid  = threadIdx.x;
    const int lane = tid & 31;
    const int w    = tid >> 5;
    const int wm   = w & 3;
    const int wn   = w >> 2;
    const int half = lane >> 4;
    const int lm   = lane & 15;
    const int row0 = blockIdx.y * BM;
    const int col0 = blockIdx.x * BN;
    const int bidx = row0 / SEQ;
    const int t0   = row0 % SEQ;

    v8f acc0 = {}, acc1 = {};

    for (int p = 0; p < 1 + KX; ++p) {
        const int shift = (p == 0) ? 0 : (p - 1);
        const float* Asrc = (p == 0) ? H : X;
        const float* Bsrc = (p == 0) ? Cg : (MT + (size_t)(p - 1) * 1024 * 1024);

        for (int kbase = 0; kbase < 1024; kbase += KC) {
            {
                const int r = tid >> 2;
                const int c = (tid & 3) << 3;
                const int trow = t0 + r - shift;
                float4 f0 = make_float4(0.f, 0.f, 0.f, 0.f);
                float4 f1 = f0;
                if (trow >= 0) {
                    const float* src =
                        Asrc + ((size_t)bidx * SEQ + trow) * 1024 + kbase + c;
                    f0 = *reinterpret_cast<const float4*>(src);
                    f1 = *reinterpret_cast<const float4*>(src + 4);
                }
                stage_A_row(s, r, c, f0, f1);
            }
            {
                const int kk = tid >> 3;
                const int c  = (tid & 7) << 3;
                const float* src = Bsrc + (size_t)(kbase + kk) * 1024 + col0 + c;
                stage_B_row(s, kk, c, *reinterpret_cast<const float4*>(src),
                                       *reinterpret_cast<const float4*>(src + 4));
            }
            __syncthreads();
            tile_mma(s, wm, wn, half, lm, acc0, acc1);
            __syncthreads();
        }
    }

    const int orow = row0 + wm * 16 + half * 8;
    const int oc0  = col0 + wn * 32 + lm;
#pragma unroll
    for (int g = 0; g < 8; ++g) {
        Og[(size_t)(orow + g) * 1024 + oc0]      = acc0[g];
        Og[(size_t)(orow + g) * 1024 + oc0 + 16] = acc1[g];
    }
}

// ===========================================================================
extern "C" void kernel_launch(void* const* d_in, const int* in_sizes, int n_in,
                              void* d_out, int out_size, void* d_ws, size_t ws_size,
                              hipStream_t stream)
{
    const float* x  = (const float*)d_in[0];   // (4,2048,1024)
    const float* h0 = (const float*)d_in[1];   // (1024,)
    const float* Ad = (const float*)d_in[2];   // (1024,)
    const float* Bm = (const float*)d_in[3];   // (1024,1024)
    const float* Cm = (const float*)d_in[4];   // (1024,1024)
    const float* Mm = (const float*)d_in[5];   // (1024,1024,10)
    float* out = (float*)d_out;

    const size_t szV  = (size_t)ROWS * 1024 * 4;        // 32 MB
    const size_t szXp = (size_t)ROWS * 1024 * 2;        // 16 MB
    const size_t szSq = (size_t)1024 * 1024 * 2;        // 2 MB
    const size_t szMp = (size_t)KX * 1024 * 1024 * 2;   // 20 MB
    const size_t need = szV + 4 * szXp + 4 * szSq + 2 * szMp;  // ~144 MB

    char* w = (char*)d_ws;

    if (ws_size >= need) {
        // ------------------- PATH A: pre-converted planes -------------------
        float* V = (float*)w;                              w += szV;
        unsigned short* Xhi  = (unsigned short*)w;         w += szXp;
        unsigned short* Xlo  = (unsigned short*)w;         w += szXp;
        unsigned short* Hhi  = (unsigned short*)w;         w += szXp;
        unsigned short* Hlo  = (unsigned short*)w;         w += szXp;
        unsigned short* BThi = (unsigned short*)w;         w += szSq;
        unsigned short* BTlo = (unsigned short*)w;         w += szSq;
        unsigned short* CThi = (unsigned short*)w;         w += szSq;
        unsigned short* CTlo = (unsigned short*)w;         w += szSq;
        unsigned short* MThi = (unsigned short*)w;         w += szMp;
        unsigned short* MTlo = (unsigned short*)w;

        conv_plane_kernel<<<dim3((ROWS * 1024 / 8) / 256), 256, 0, stream>>>(x, Xhi, Xlo);
        conv_T_kernel<<<dim3((1024 * 128) / 256), 256, 0, stream>>>(Bm, BThi, BTlo);
        conv_T_kernel<<<dim3((1024 * 128) / 256), 256, 0, stream>>>(Cm, CThi, CTlo);
        conv_M_kernel<<<dim3((1024 * 128) / 256), 256, 0, stream>>>(Mm, MThi, MTlo);
        gemm_pre<<<dim3(STATE / 64, ROWS / 128), 256, 0, stream>>>(Xhi, Xlo, BThi, BTlo, V);
        scan_split_kernel<<<dim3(4096 / 256), 256, 0, stream>>>(V, Ad, h0, Hhi, Hlo);
        fused_pre<<<dim3(DOUT / 64, ROWS / 128), 256, 0, stream>>>(
            Hhi, Hlo, CThi, CTlo, Xhi, Xlo, MThi, MTlo, out);
    } else {
        // ------------------- PATH B: convert-in-kernel ----------------------
        float* V  = (float*)w;
        float* MT = (float*)(w + szV);

        transpose_M_kernel<<<dim3((1024 * 1024) / 256), 256, 0, stream>>>(Mm, MT);
        gemm_bf16s_wmma<<<dim3(STATE / BN, ROWS / BM), 256, 0, stream>>>(x, Bm, V);
        scan_kernel<<<dim3(4096 / 256), 256, 0, stream>>>(V, Ad, h0);
        fused_out_wmma<<<dim3(DOUT / BN, ROWS / BM), 256, 0, stream>>>(V, Cm, x, MT, out);
    }
}
